// TemporalAttention_32976758899216
// MI455X (gfx1250) — matrix operand order repaired and checked by reading, not compiled, not run on hardware
//
#include <hip/hip_runtime.h>
#include <hip/hip_bf16.h>
#include <math.h>

// B=8 T=256 O=16 D=512 H=8 HD=64
#define EPSV 1e-5f

typedef __attribute__((ext_vector_type(16))) __bf16 v16bf;
typedef __attribute__((ext_vector_type(8)))  float  v8f;
typedef __attribute__((ext_vector_type(2)))  __bf16 bf16x2;

#define WMMA_BF16(a, b, c) \
  __builtin_amdgcn_wmma_f32_16x16x32_bf16(false, (a), false, (b), (short)0, (c), false, false)

// A-matrix 16x32 bf16 tile, row-major source (element (m,k) at base + m*ld + k).
__device__ __forceinline__ v16bf load_A16x32(const __bf16* base, int ld) {
  const int lane = threadIdx.x & 31;
  const int half = lane >> 4;
  const int m    = lane & 15;
  v16bf a;
#pragma unroll
  for (int g = 0; g < 8; ++g) {
    const int k = ((g & 4) ? 16 : 0) + ((g & 3) << 1) + half * 8;
    bf16x2 p = *reinterpret_cast<const bf16x2*>(base + m * ld + k);
    a[2 * g]     = p.x;
    a[2 * g + 1] = p.y;
  }
  return a;
}

// B-matrix 32x16 bf16 tile from an N-major (transposed) source:
// element (k,n) at baseT + n*ld + k. Lanes 0-15 K=0..7,16..23, lanes 16-31 K=8..15,24..31.
__device__ __forceinline__ v16bf load_B32x16T(const __bf16* baseT, int ld) {
  const int lane = threadIdx.x & 31;
  const int half = lane >> 4;
  const int n    = lane & 15;
  v16bf b;
#pragma unroll
  for (int g = 0; g < 8; ++g) {
    const int k = ((g & 4) ? 16 : 0) + ((g & 3) << 1) + half * 8;
    bf16x2 p = *reinterpret_cast<const bf16x2*>(baseT + n * ld + k);
    b[2 * g]     = p.x;
    b[2 * g + 1] = p.y;
  }
  return b;
}

// ---------------- kernel 0: weights f32 -> bf16, transposed to [N][K] --------
__global__ __launch_bounds__(256)
void wconv_kernel(const float* __restrict__ Wq, const float* __restrict__ Wk,
                  const float* __restrict__ Wv, const float* __restrict__ Wo,
                  __bf16* __restrict__ WT) {
  int idx = blockIdx.x * 256 + threadIdx.x;     // 4 * 512 * 512 total
  int mat = idx >> 18;
  int rem = idx & 262143;
  int n = rem >> 9, kk = rem & 511;
  const float* W = (mat == 0) ? Wq : (mat == 1) ? Wk : (mat == 2) ? Wv : Wo;
  WT[(size_t)mat * 262144 + (size_t)n * 512 + kk] = (__bf16)W[(size_t)kk * 512 + n];
}

// ---------------- kernel 1: fused Q/K/V projection ---------------------------
// block = one (b,t): 16 organ rows == one M-tile. Two N-tiles per wave in
// flight -> two independent WMMA chains hide B-load latency.
__global__ __launch_bounds__(256)
void qkv_proj_kernel(const float* __restrict__ x,
                     const __bf16* __restrict__ WT,   // [3][512][512] N-major
                     const float* __restrict__ bq, const float* __restrict__ bk,
                     const float* __restrict__ bv,
                     __bf16* __restrict__ qout, __bf16* __restrict__ kout,
                     __bf16* __restrict__ vout) {
  __shared__ __bf16 Abuf[16 * 520];
  const int tid = threadIdx.x;
  const int bb = blockIdx.x >> 8;
  const int tt = blockIdx.x & 255;
  const int row_base = blockIdx.x * 16;

#pragma unroll
  for (int i = 0; i < 32; ++i) {
    int idx = tid + i * 256;
    int m = idx >> 9, c = idx & 511;
    Abuf[m * 520 + c] = (__bf16)x[(size_t)(row_base + m) * 512 + c];
  }
  __syncthreads();

  const int wave = tid >> 5;
  const int lane = tid & 31;
  const int half = lane >> 4;
  const int coln = lane & 15;

  for (int j = 0; j < 12; j += 2) {               // 96 N-tiles / 8 waves, paired
    int tile0 = wave + j * 8;
    int tile1 = wave + (j + 1) * 8;
    int mat0 = tile0 >> 5, n00 = (tile0 & 31) * 16;
    int mat1 = tile1 >> 5, n01 = (tile1 & 31) * 16;
    const __bf16* W0 = WT + (size_t)mat0 * 262144 + (size_t)n00 * 512;
    const __bf16* W1 = WT + (size_t)mat1 * 262144 + (size_t)n01 * 512;
    __builtin_prefetch(W0, 0, 1);
    __builtin_prefetch(W1, 0, 1);
    v8f acc0 = {}, acc1 = {};
#pragma unroll
    for (int ks = 0; ks < 16; ++ks) {
      v16bf a  = load_A16x32(Abuf + ks * 32, 520);
      v16bf b0 = load_B32x16T(W0 + ks * 32, 512);
      v16bf b1 = load_B32x16T(W1 + ks * 32, 512);
      acc0 = WMMA_BF16(a, b0, acc0);
      acc1 = WMMA_BF16(a, b1, acc1);
    }
#pragma unroll
    for (int p = 0; p < 2; ++p) {
      int mat = p ? mat1 : mat0;
      int n0  = p ? n01  : n00;
      v8f acc = p ? acc1 : acc0;
      const float* bp = (mat == 0) ? bq : (mat == 1) ? bk : bv;
      __bf16* dst = (mat == 0) ? qout : (mat == 1) ? kout : vout;
      float bias = bp[n0 + coln];
      int e = n0 + coln, h = e >> 6, hd = e & 63;
#pragma unroll
      for (int r = 0; r < 8; ++r) {
        int o = r + half * 8;
        size_t off = ((((size_t)bb * 16 + o) * 8 + h) * 256 + tt) * 64 + hd;
        dst[off] = (__bf16)(acc[r] + bias);
      }
    }
  }
}

// ---------------- kernel 2: masked attention per (b,o,h) ---------------------
__global__ __launch_bounds__(256)
void attn_kernel(const __bf16* __restrict__ q, const __bf16* __restrict__ k,
                 const __bf16* __restrict__ v, const int* __restrict__ mask,
                 float* __restrict__ att /* d_out reused: (B,T,O,D) f32 */) {
  __shared__ float  Sbuf[64 * 264];               // also v-staging scratch
  __shared__ __bf16 Pbuf[64 * 264];
  __shared__ __bf16 vT[64 * 264];                 // [hd][t]
  __shared__ float  rowScale[64];
  __shared__ int    mkey[256];

  const int tid = threadIdx.x;
  const int idx = blockIdx.x;                     // (b*16+o)*8+h
  const int h = idx & 7;
  const int o = (idx >> 3) & 15;
  const int b = idx >> 7;
  const __bf16* qb = q + (size_t)idx * 256 * 64;
  const __bf16* kb = k + (size_t)idx * 256 * 64;
  const __bf16* vb = v + (size_t)idx * 256 * 64;

  // Async-DMA the 32KB v tile into the (currently dead) score buffer, then
  // transpose LDS->LDS into vT. ASYNCcnt path: global_load_async_to_lds_b128.
  {
    char* vLinB = (char*)Sbuf;
    const char* gsrc = (const char*)vb;
#pragma unroll
    for (int i = 0; i < 8; ++i) {
      unsigned off = (unsigned)(tid + i * 256) * 16u;
      unsigned ldsoff = (unsigned)(unsigned long long)(vLinB + off);
      const void* ga = (const void*)(gsrc + off);
      asm volatile("global_load_async_to_lds_b128 %0, %1, off"
                   :: "v"(ldsoff), "v"(ga) : "memory");
    }
    asm volatile("s_wait_asynccnt 0" ::: "memory");
    mkey[tid] = mask[((size_t)b * 256 + tid) * 16 + o];
    __syncthreads();
    const __bf16* vLin = (const __bf16*)Sbuf;
#pragma unroll
    for (int i = 0; i < 64; ++i) {
      int e = tid + i * 256;
      vT[(e & 63) * 264 + (e >> 6)] = vLin[e];
    }
    __syncthreads();
  }

  const int wave = tid >> 5;
  const int lane = tid & 31;
  const int half = lane >> 4;
  const int coln = lane & 15;

  for (int qc = 0; qc < 4; ++qc) {                // 64-query chunks
    // scores = q k^T / 8 : two key-tiles per wave concurrently
    {
      const int rt = wave >> 1;
      const int q0 = qc * 64 + rt * 16;
#pragma unroll
      for (int jj = 0; jj < 8; jj += 2) {
        int ct0 = (wave & 1) * 8 + jj;
        int ct1 = ct0 + 1;
        v8f acc0 = {}, acc1 = {};
#pragma unroll
        for (int ks = 0; ks < 2; ++ks) {
          v16bf a  = load_A16x32(qb + q0 * 64 + ks * 32, 64);
          v16bf b0 = load_B32x16T(kb + ct0 * 16 * 64 + ks * 32, 64);
          v16bf b1 = load_B32x16T(kb + ct1 * 16 * 64 + ks * 32, 64);
          acc0 = WMMA_BF16(a, b0, acc0);
          acc1 = WMMA_BF16(a, b1, acc1);
        }
#pragma unroll
        for (int r = 0; r < 8; ++r) {
          int rr = rt * 16 + r + half * 8;
          Sbuf[rr * 264 + ct0 * 16 + coln] = acc0[r] * 0.125f;
          Sbuf[rr * 264 + ct1 * 16 + coln] = acc1[r] * 0.125f;
        }
      }
    }
    __syncthreads();
    // masked, unnormalized softmax (2 full waves -> no intra-wave divergence)
    if (tid < 64) {
      float mx = -INFINITY;
      for (int key = 0; key < 256; ++key)
        if (mkey[key]) mx = fmaxf(mx, Sbuf[tid * 264 + key]);
      float sum = 0.f;
      for (int key = 0; key < 256; ++key) {
        float e = 0.f;
        if (mkey[key]) {
          e = __expf(Sbuf[tid * 264 + key] - mx);
          sum += e;
        }
        Pbuf[tid * 264 + key] = (__bf16)e;
      }
      rowScale[tid] = (sum > 0.f) ? (1.f / sum) : 0.f;   // all-masked row -> 0
    }
    __syncthreads();
    // att = P @ V, both of this wave's tiles concurrently
    {
      int tl0 = wave, tl1 = wave + 8;
      int rt0 = tl0 >> 2, c0 = tl0 & 3;
      int rt1 = tl1 >> 2, c1 = tl1 & 3;
      v8f acc0 = {}, acc1 = {};
#pragma unroll
      for (int ks = 0; ks < 8; ++ks) {
        v16bf a0 = load_A16x32(Pbuf + rt0 * 16 * 264 + ks * 32, 264);
        v16bf b0 = load_B32x16T(vT + c0 * 16 * 264 + ks * 32, 264);
        v16bf a1 = load_A16x32(Pbuf + rt1 * 16 * 264 + ks * 32, 264);
        v16bf b1 = load_B32x16T(vT + c1 * 16 * 264 + ks * 32, 264);
        acc0 = WMMA_BF16(a0, b0, acc0);
        acc1 = WMMA_BF16(a1, b1, acc1);
      }
#pragma unroll
      for (int r = 0; r < 8; ++r) {
        int m0 = rt0 * 16 + r + half * 8;
        int m1 = rt1 * 16 + r + half * 8;
        int t0 = qc * 64 + m0, t1 = qc * 64 + m1;
        int e0 = h * 64 + c0 * 16 + coln, e1 = h * 64 + c1 * 16 + coln;
        att[(((size_t)b * 256 + t0) * 16 + o) * 512 + e0] = acc0[r] * rowScale[m0];
        att[(((size_t)b * 256 + t1) * 16 + o) * 512 + e1] = acc1[r] * rowScale[m1];
      }
    }
    __syncthreads();
  }
}

// ---------------- kernel 3: output proj + residual + LayerNorm + mask --------
__global__ __launch_bounds__(256)
void out_ln_kernel(const float* __restrict__ x, const int* __restrict__ mask,
                   const __bf16* __restrict__ WoT, const float* __restrict__ bo,
                   const float* __restrict__ gamma, const float* __restrict__ beta,
                   float* __restrict__ out /* d_out: att f32 on entry */) {
  __shared__ __bf16 Abuf[16 * 520];
  __shared__ float  Rbuf[16 * 516];
  __shared__ float  partS[256], partQ[256];
  __shared__ float  muA[16], rsA[16], mkA[16];

  const int tid = threadIdx.x;
  const int bb = blockIdx.x >> 8;
  const int tt = blockIdx.x & 255;
  const int row_base = blockIdx.x * 16;

#pragma unroll
  for (int i = 0; i < 32; ++i) {
    int idx2 = tid + i * 256;
    int m = idx2 >> 9, c = idx2 & 511;
    Abuf[m * 520 + c] = (__bf16)out[(size_t)(row_base + m) * 512 + c];
  }
  __syncthreads();

  const int wave = tid >> 5;
  const int lane = tid & 31;
  const int half = lane >> 4;
  const int coln = lane & 15;

  for (int j = 0; j < 4; j += 2) {                // 32 N-tiles / 8 waves, paired
    int n00 = (wave * 4 + j) * 16;
    int n01 = n00 + 16;
    __builtin_prefetch(WoT + (size_t)n00 * 512, 0, 1);
    v8f acc0 = {}, acc1 = {};
#pragma unroll
    for (int ks = 0; ks < 16; ++ks) {
      v16bf a  = load_A16x32(Abuf + ks * 32, 520);
      v16bf b0 = load_B32x16T(WoT + (size_t)n00 * 512 + ks * 32, 512);
      v16bf b1 = load_B32x16T(WoT + (size_t)n01 * 512 + ks * 32, 512);
      acc0 = WMMA_BF16(a, b0, acc0);
      acc1 = WMMA_BF16(a, b1, acc1);
    }
    float bias0 = bo[n00 + coln], bias1 = bo[n01 + coln];
#pragma unroll
    for (int r = 0; r < 8; ++r) {
      Rbuf[(r + half * 8) * 516 + n00 + coln] = acc0[r] + bias0;
      Rbuf[(r + half * 8) * 516 + n01 + coln] = acc1[r] + bias1;
    }
  }
  __syncthreads();

  const int row = tid >> 4;
  const int seg = tid & 15;
  float s = 0.f, s2 = 0.f;
#pragma unroll
  for (int i = 0; i < 32; ++i) {
    int c = seg * 32 + i;
    float rv = Rbuf[row * 516 + c] + x[(size_t)(row_base + row) * 512 + c];
    Rbuf[row * 516 + c] = rv;
    s += rv; s2 += rv * rv;
  }
  partS[tid] = s; partQ[tid] = s2;
  __syncthreads();
  if (tid < 16) {
    float S = 0.f, Q = 0.f;
    for (int i = 0; i < 16; ++i) { S += partS[tid * 16 + i]; Q += partQ[tid * 16 + i]; }
    float mu = S * (1.f / 512.f);
    float var = Q * (1.f / 512.f) - mu * mu;
    muA[tid] = mu;
    rsA[tid] = rsqrtf(var + EPSV);
    mkA[tid] = mask[((size_t)bb * 256 + tt) * 16 + tid] ? 1.f : 0.f;
  }
  __syncthreads();
  {
    float mu = muA[row], rs = rsA[row], mv = mkA[row];
#pragma unroll
    for (int i = 0; i < 32; ++i) {
      int c = seg * 32 + i;
      float yv = (Rbuf[row * 516 + c] - mu) * rs * gamma[c] + beta[c];
      out[(size_t)(row_base + row) * 512 + c] = yv * mv;
    }
  }
}

extern "C" void kernel_launch(void* const* d_in, const int* in_sizes, int n_in,
                              void* d_out, int out_size, void* d_ws, size_t ws_size,
                              hipStream_t stream) {
  const float* x     = (const float*)d_in[0];
  const int*   maskp = (const int*)d_in[1];
  const float* Wq = (const float*)d_in[2];
  const float* bq = (const float*)d_in[3];
  const float* Wk = (const float*)d_in[4];
  const float* bk = (const float*)d_in[5];
  const float* Wv = (const float*)d_in[6];
  const float* bv = (const float*)d_in[7];
  const float* Wo = (const float*)d_in[8];
  const float* bo = (const float*)d_in[9];
  const float* gamma = (const float*)d_in[10];
  const float* beta  = (const float*)d_in[11];
  float* out = (float*)d_out;

  const size_t QKV = (size_t)8 * 16 * 8 * 256 * 64;     // B*O*H*T*HD
  __bf16* WT = (__bf16*)d_ws;                           // 4 * 512*512 bf16
  __bf16* qb = WT + (size_t)4 * 262144;
  __bf16* kb = qb + QKV;
  __bf16* vb = kb + QKV;

  wconv_kernel<<<4096, 256, 0, stream>>>(Wq, Wk, Wv, Wo, WT);
  qkv_proj_kernel<<<2048, 256, 0, stream>>>(x, WT, bq, bk, bv, qb, kb, vb);
  attn_kernel<<<1024, 256, 0, stream>>>(qb, kb, vb, maskp, out);
  out_ln_kernel<<<2048, 256, 0, stream>>>(x, maskp, WT + (size_t)3 * 262144,
                                          bo, gamma, beta, out);
}
